// RoPEAttention_19619410608645
// MI455X (gfx1250) — matrix operand order repaired and checked by reading, not compiled, not run on hardware
//
#include <hip/hip_runtime.h>
#include <stdint.h>

// RoPE attention for MI455X (gfx1250): all GEMMs on v_wmma_f32_16x16x32_bf16,
// K/V tiles staged into LDS via the Tensor Data Mover (tensor_load_to_lds).
// B=4, L=2048, D=1024, H=16, HD=64.

#define B_ 4
#define L_ 2048
#define D_ 1024
#define H_ 16
#define HD_ 64

typedef __attribute__((ext_vector_type(16))) __bf16 bf16x16;
typedef __attribute__((ext_vector_type(8)))  __bf16 bf16x8;
typedef __attribute__((ext_vector_type(4)))  __bf16 bf16x4;
typedef __attribute__((ext_vector_type(8)))  float  f32x8;
typedef __attribute__((ext_vector_type(4)))  float  f32x4;
typedef unsigned int u32x4 __attribute__((ext_vector_type(4)));
typedef int          i32x4 __attribute__((ext_vector_type(4)));
typedef int          i32x8 __attribute__((ext_vector_type(8)));

__device__ __forceinline__ bf16x16 cat8(bf16x8 lo, bf16x8 hi) {
  return __builtin_shufflevector(lo, hi, 0,1,2,3,4,5,6,7,8,9,10,11,12,13,14,15);
}

// A-fragment (16x32 bf16, wave32): for lane-half h,
// elements 0..15 hold K = 16h..16h+15 (same order as B-fragment).
__device__ __forceinline__ bf16x16 load_afrag(const __bf16* p, int half) {
  bf16x8 lo = *(const bf16x8*)(p + half * 16);
  bf16x8 hi = *(const bf16x8*)(p + half * 16 + 8);
  return cat8(lo, hi);
}
// B-fragment (32x16 bf16): lane ln = column, K = 16*half + e; p includes half*16.
__device__ __forceinline__ bf16x16 load_bfrag(const __bf16* p) {
  bf16x8 lo = *(const bf16x8*)(p);
  bf16x8 hi = *(const bf16x8*)(p + 8);
  return cat8(lo, hi);
}

__device__ __forceinline__ float rowmax16(float v) {
  #pragma unroll
  for (int off = 1; off < 16; off <<= 1) v = fmaxf(v, __shfl_xor(v, off, 16));
  return v;
}
__device__ __forceinline__ float rowsum16(float v) {
  #pragma unroll
  for (int off = 1; off < 16; off <<= 1) v += __shfl_xor(v, off, 16);
  return v;
}

// ---------------------------------------------------------------------------
// TDM: issue a 2D bf16 tile load (tile_w x tile_h) global -> LDS.
// D# packing per CDNA5 ISA 8.3/8.4: group0 = {count, lds_addr, gaddr, type=2},
// group1 = {data_size=2B, tensor dims, tile dims, dim0 stride}. 2D -> rest 0.
// This toolchain exposes the 6-arg builtin:
//   (u32x4 g0, i32x8 g1, i32x4, i32x4, i32x8, i32 cpol)
// ---------------------------------------------------------------------------
#if __has_builtin(__builtin_amdgcn_tensor_load_to_lds) && \
    __has_builtin(__builtin_amdgcn_s_wait_tensorcnt)
#define USE_TDM 1
#else
#define USE_TDM 0
#endif

__device__ __forceinline__ void tdm_load_2d_bf16(
    uint32_t lds_addr, const void* gaddr,
    uint32_t tensor_w, uint32_t tensor_h,
    uint32_t tile_w, uint32_t tile_h, uint32_t stride_elems)
{
#if USE_TDM
  const uint64_t ga = (uint64_t)(uintptr_t)gaddr;
  u32x4 g0;
  g0[0] = 1u;                                               // count=1, no gather
  g0[1] = lds_addr;                                         // lds_addr[31:0]
  g0[2] = (uint32_t)ga;                                     // global_addr[31:0]
  g0[3] = (uint32_t)((ga >> 32) & 0x1FFFFFFu) | (2u << 30); // addr[56:32] | type=2
  i32x8 g1;
  g1[0] = (int)(1u << 16);                                  // data_size=1 (2 bytes)
  g1[1] = (int)((tensor_w & 0xFFFFu) << 16);                // tensor_dim0[15:0]
  g1[2] = (int)(((tensor_w >> 16) & 0xFFFFu) |              // tensor_dim0[31:16]
                ((tensor_h & 0xFFFFu) << 16));              // tensor_dim1[15:0]
  g1[3] = (int)(((tensor_h >> 16) & 0xFFFFu) |              // tensor_dim1[31:16]
                ((tile_w & 0xFFFFu) << 16));                // tile_dim0
  g1[4] = (int)(tile_h & 0xFFFFu);                          // tile_dim1 (tile_dim2=0)
  g1[5] = (int)stride_elems;                                // tensor_dim0_stride[31:0]
  g1[6] = 0; g1[7] = 0;
  i32x4 g2 = {0, 0, 0, 0}, g3 = {0, 0, 0, 0};
  i32x8 g4 = {0, 0, 0, 0, 0, 0, 0, 0};                      // unused (2D tensor)
  __builtin_amdgcn_tensor_load_to_lds(g0, g1, g2, g3, g4, 0);
#else
  (void)lds_addr; (void)gaddr; (void)tensor_w; (void)tensor_h;
  (void)tile_w; (void)tile_h; (void)stride_elems;
#endif
}

// ---------------------------------------------------------------------------
// Kernel 0: one-shot fp32 -> bf16 conversion (vectorized 4-wide).
// ---------------------------------------------------------------------------
__global__ __launch_bounds__(256) void cvt_f32_bf16_kernel(
    const float* __restrict__ src, __bf16* __restrict__ dst, int n4)
{
  const int i = blockIdx.x * 256 + threadIdx.x;
  if (i < n4) {
    f32x4 v = *(const f32x4*)(src + (size_t)i * 4);
    bf16x4 o;
    o[0] = (__bf16)v[0]; o[1] = (__bf16)v[1];
    o[2] = (__bf16)v[2]; o[3] = (__bf16)v[3];
    *(bf16x4*)(dst + (size_t)i * 4) = o;
  }
}

// ---------------------------------------------------------------------------
// Kernel 1: y = x @ W^T + b, then RoPE (for q,k). Paired 16x16 tiles so
// rotate-half (d <-> d+32) stays in-register.
// Q,K -> bf16 (B,H,L,HD); V -> bf16 transposed (B,H,HD,L).
// ---------------------------------------------------------------------------
__global__ __launch_bounds__(32) void qkv_rope_kernel(
    const __bf16* __restrict__ Xq, const __bf16* __restrict__ Xk,
    const __bf16* __restrict__ Xv,
    const __bf16* __restrict__ Wqb, const __bf16* __restrict__ Wkb,
    const __bf16* __restrict__ Wvb,
    const float* __restrict__ bq, const float* __restrict__ bk,
    const float* __restrict__ bv,
    const float* __restrict__ cosT, const float* __restrict__ sinT,
    __bf16* __restrict__ Qws, __bf16* __restrict__ Kws,
    __bf16* __restrict__ Vws)
{
  const int lane = threadIdx.x;
  const int half = lane >> 4;
  const int ln   = lane & 15;
  const int sel  = blockIdx.z;          // 0=q, 1=k, 2=v
  const int row0 = blockIdx.x * 16;
  const int h    = blockIdx.y >> 1;
  const int s    = blockIdx.y & 1;
  const int n0   = h * 64 + s * 16;     // paired with n0+32

  const __bf16* x    = (sel == 0) ? Xq  : (sel == 1) ? Xk  : Xv;
  const __bf16* W    = (sel == 0) ? Wqb : (sel == 1) ? Wkb : Wvb;
  const float*  bias = (sel == 0) ? bq  : (sel == 1) ? bk  : bv;

  const __bf16* xrow  = x + (size_t)(row0 + ln) * D_;
  const __bf16* wrow0 = W + (size_t)(n0 + ln) * D_ + half * 16;
  const __bf16* wrow1 = W + (size_t)(n0 + 32 + ln) * D_ + half * 16;

  f32x8 acc0 = {}, acc1 = {};
  for (int k0 = 0; k0 < D_; k0 += 32) {
    __builtin_prefetch(wrow0 + k0 + 512, 0, 3);
    __builtin_prefetch(wrow1 + k0 + 512, 0, 3);
    const bf16x16 a  = load_afrag(xrow + k0, half);
    const bf16x16 b0 = load_bfrag(wrow0 + k0);
    const bf16x16 b1 = load_bfrag(wrow1 + k0);
    acc0 = __builtin_amdgcn_wmma_f32_16x16x32_bf16(false, a, false, b0,
                                                   (short)0, acc0, false, false);
    acc1 = __builtin_amdgcn_wmma_f32_16x16x32_bf16(false, a, false, b1,
                                                   (short)0, acc1, false, false);
  }

  const float bias0 = bias[n0 + ln];
  const float bias1 = bias[n0 + 32 + ln];
  const int dl = s * 16 + ln;

  #pragma unroll
  for (int r = 0; r < 8; ++r) {
    const int m     = half * 8 + r;
    const int row   = row0 + m;
    const int b_idx = row >> 11;
    const int l_pos = row & (L_ - 1);
    const float y0 = acc0[r] + bias0;
    const float y1 = acc1[r] + bias1;
    if (sel < 2) {
      const float* cr = cosT + (size_t)l_pos * HD_;
      const float* sr = sinT + (size_t)l_pos * HD_;
      const float o0 = y0 * cr[dl]      - y1 * sr[dl];
      const float o1 = y1 * cr[dl + 32] + y0 * sr[dl + 32];
      __bf16* dst = (sel == 0 ? Qws : Kws) +
                    ((size_t)(b_idx * H_ + h) * L_ + l_pos) * HD_;
      dst[dl]      = (__bf16)o0;
      dst[dl + 32] = (__bf16)o1;
    } else {
      __bf16* dst = Vws + (size_t)(b_idx * H_ + h) * HD_ * L_;
      dst[(size_t)dl * L_ + l_pos]        = (__bf16)y0;
      dst[(size_t)(dl + 32) * L_ + l_pos] = (__bf16)y1;
    }
  }
}

// ---------------------------------------------------------------------------
// Kernel 2: flash attention. Block = 8 waves = 128 queries of one (b,h).
// Wave 0 TDM-stages the shared 32-key K tile (32x64) and V^T tile (64x32)
// into LDS each iteration (s_wait_tensorcnt + workgroup barrier), then each
// wave runs 4 WMMA for S, an online softmax, and 4 WMMA for P@V.
// Waves past their causal limit skip compute (wave-uniform branch; the
// barriers sit at points every wave reaches).
// ---------------------------------------------------------------------------
__global__ __launch_bounds__(256) void flash_attn_kernel(
    const __bf16* __restrict__ Qws, const __bf16* __restrict__ Kws,
    const __bf16* __restrict__ Vws, __bf16* __restrict__ Ows,
    const int* __restrict__ causal_p)
{
  __shared__ __bf16 sK[32 * HD_];        // 32 keys x 64  (4 KB)
  __shared__ __bf16 sV[HD_ * 32];        // 64 d   x 32 keys (4 KB)
  __shared__ __bf16 sPall[8 * 16 * 32];  // per-wave P staging (8 KB)

  const int tid  = threadIdx.x;
  const int lane = tid & 31;
  const int wave = tid >> 5;
  const int half = lane >> 4;
  const int ln   = lane & 15;
  const int qi   = blockIdx.x * 8 + wave;   // this wave's 16-query tile
  const int h    = blockIdx.y;
  const int b    = blockIdx.z;
  const int causal = causal_p[0];

  const size_t bh = (size_t)(b * H_ + h);
  const __bf16* Q = Qws + bh * (size_t)L_ * HD_;
  const __bf16* K = Kws + bh * (size_t)L_ * HD_;
  const __bf16* V = Vws + bh * (size_t)HD_ * L_;     // transposed (HD, L)
  __bf16* sP = sPall + wave * (16 * 32);

  const __bf16* qrow = Q + (size_t)(qi * 16 + ln) * HD_;
  const bf16x16 aq0 = load_afrag(qrow, half);
  const bf16x16 aq1 = load_afrag(qrow + 32, half);

  float m_run[8], l_run[8];
  #pragma unroll
  for (int r = 0; r < 8; ++r) { m_run[r] = -3.0e38f; l_run[r] = 0.0f; }
  f32x8 o0 = {}, o1 = {}, o2 = {}, o3 = {};

  // Block-level key range: last q-tile of the block dictates the bound.
  const int jmax = causal ? ((blockIdx.x * 128 + 127) >> 5) : (L_ / 32 - 1);
  for (int j = 0; j <= jmax; ++j) {
    const int key0 = j * 32;

    __syncthreads();                 // previous tiles fully consumed
    if (wave == 0) {
#if USE_TDM
      // K tile: 32 rows x 64 cols, row-major, contiguous (stride 64).
      tdm_load_2d_bf16((uint32_t)(uintptr_t)(void*)sK,
                       K + (size_t)key0 * HD_, HD_, L_, HD_, 32, HD_);
      // V^T tile: 64 rows x 32 cols from (HD, L) layout (stride L).
      tdm_load_2d_bf16((uint32_t)(uintptr_t)(void*)sV,
                       V + key0, L_, HD_, 32, HD_, L_);
      __builtin_amdgcn_s_wait_tensorcnt(0);
#else
      // Fallback: manual staging by wave 0 (b128 copies).
      const __bf16* kg = K + (size_t)key0 * HD_;   // contiguous 2048 elems
      for (int t = lane; t < (32 * HD_) / 8; t += 32)
        ((bf16x8*)sK)[t] = *(const bf16x8*)(kg + t * 8);
      for (int t = lane; t < (HD_ * 32) / 8; t += 32) {
        const int rr = t >> 2, cc = t & 3;
        ((bf16x8*)sV)[t] = *(const bf16x8*)(V + (size_t)rr * L_ + key0 + cc * 8);
      }
#endif
    }
    __syncthreads();                 // tiles visible to all waves

    const bool active = (!causal) || (key0 <= qi * 16 + 15);
    if (active) {
      f32x8 s0 = {}, s1 = {};
      {
        const __bf16* krow = sK + (size_t)ln * HD_ + half * 16;
        s0 = __builtin_amdgcn_wmma_f32_16x16x32_bf16(false, aq0, false,
               load_bfrag(krow),      (short)0, s0, false, false);
        s0 = __builtin_amdgcn_wmma_f32_16x16x32_bf16(false, aq1, false,
               load_bfrag(krow + 32), (short)0, s0, false, false);
      }
      {
        const __bf16* krow = sK + (size_t)(16 + ln) * HD_ + half * 16;
        s1 = __builtin_amdgcn_wmma_f32_16x16x32_bf16(false, aq0, false,
               load_bfrag(krow),      (short)0, s1, false, false);
        s1 = __builtin_amdgcn_wmma_f32_16x16x32_bf16(false, aq1, false,
               load_bfrag(krow + 32), (short)0, s1, false, false);
      }

      #pragma unroll
      for (int r = 0; r < 8; ++r) {
        const int mg = qi * 16 + half * 8 + r;
        float v0 = s0[r] * 0.125f;               // 1/sqrt(64)
        float v1 = s1[r] * 0.125f;
        if (causal) {
          if (key0 + ln > mg)      v0 = -3.0e38f;
          if (key0 + 16 + ln > mg) v1 = -3.0e38f;
        }
        const float rm    = rowmax16(fmaxf(v0, v1));
        const float mnew  = fmaxf(m_run[r], rm);
        const float alpha = __expf(m_run[r] - mnew);
        const float p0 = __expf(v0 - mnew);
        const float p1 = __expf(v1 - mnew);
        const float ls = rowsum16(p0 + p1);
        l_run[r] = l_run[r] * alpha + ls;
        m_run[r] = mnew;
        o0[r] *= alpha; o1[r] *= alpha; o2[r] *= alpha; o3[r] *= alpha;
        const int m = half * 8 + r;
        sP[m * 32 + ln]      = (__bf16)p0;
        sP[m * 32 + 16 + ln] = (__bf16)p1;
      }
      // In-wave LDS ordering: DS ops of one wave execute in order; the asm
      // fence stops the compiler from reordering the reads above the writes.
      asm volatile("" ::: "memory");
      const bf16x16 ap = load_afrag(sP + ln * 32, half);
      asm volatile("" ::: "memory");

      #pragma unroll
      for (int c = 0; c < 4; ++c) {
        const __bf16* vrow = sV + (size_t)(c * 16 + ln) * 32 + half * 16;
        const bf16x16 bv = load_bfrag(vrow);
        f32x8& oc = (c == 0) ? o0 : (c == 1) ? o1 : (c == 2) ? o2 : o3;
        oc = __builtin_amdgcn_wmma_f32_16x16x32_bf16(false, ap, false, bv,
                                                     (short)0, oc, false, false);
      }
    }
  }

  // normalize and store attn output as bf16 (B, L, D)
  #pragma unroll
  for (int r = 0; r < 8; ++r) {
    const int m     = half * 8 + r;
    const int l_pos = qi * 16 + m;
    const float inv = 1.0f / l_run[r];
    __bf16* orow = Ows + ((size_t)b * L_ + l_pos) * D_ + h * HD_ + ln;
    orow[0]  = (__bf16)(o0[r] * inv);
    orow[16] = (__bf16)(o1[r] * inv);
    orow[32] = (__bf16)(o2[r] * inv);
    orow[48] = (__bf16)(o3[r] * inv);
  }
}

// ---------------------------------------------------------------------------
// Kernel 3: out = O @ Wo^T + bo, fp32 output. One wave per 16x16 tile.
// ---------------------------------------------------------------------------
__global__ __launch_bounds__(32) void out_proj_kernel(
    const __bf16* __restrict__ Ows, const __bf16* __restrict__ Wob,
    const float* __restrict__ bo, float* __restrict__ out)
{
  const int lane = threadIdx.x;
  const int half = lane >> 4;
  const int ln   = lane & 15;
  const int row0 = blockIdx.x * 16;
  const int n0   = blockIdx.y * 16;

  const __bf16* arow = Ows + (size_t)(row0 + ln) * D_;
  const __bf16* wrow = Wob + (size_t)(n0 + ln) * D_ + half * 16;

  f32x8 acc = {};
  for (int k0 = 0; k0 < D_; k0 += 32) {
    __builtin_prefetch(wrow + k0 + 512, 0, 3);
    const bf16x16 a  = load_afrag(arow + k0, half);
    const bf16x16 bb = load_bfrag(wrow + k0);
    acc = __builtin_amdgcn_wmma_f32_16x16x32_bf16(false, a, false, bb,
                                                  (short)0, acc, false, false);
  }
  const float bias = bo[n0 + ln];
  #pragma unroll
  for (int r = 0; r < 8; ++r) {
    const int m = half * 8 + r;
    out[(size_t)(row0 + m) * D_ + n0 + ln] = acc[r] + bias;
  }
}

// ---------------------------------------------------------------------------
extern "C" void kernel_launch(void* const* d_in, const int* in_sizes, int n_in,
                              void* d_out, int out_size, void* d_ws, size_t ws_size,
                              hipStream_t stream) {
  (void)in_sizes; (void)n_in; (void)out_size; (void)ws_size;
  const float* q_in = (const float*)d_in[0];
  const float* k_in = (const float*)d_in[1];
  const float* v_in = (const float*)d_in[2];
  const float* cosT = (const float*)d_in[3];
  const float* sinT = (const float*)d_in[4];
  const float* Wq   = (const float*)d_in[5];
  const float* bq   = (const float*)d_in[6];
  const float* Wk   = (const float*)d_in[7];
  const float* bk   = (const float*)d_in[8];
  const float* Wv   = (const float*)d_in[9];
  const float* bv   = (const float*)d_in[10];
  const float* Wo   = (const float*)d_in[11];
  const float* bo   = (const float*)d_in[12];
  const int* causal = (const int*)d_in[13];

  // Workspace (bf16): 3x act (B*L*D) + 4x weight (D*D) + Q,K,Vt,O ~= 126 MB.
  const size_t BLD = (size_t)B_ * L_ * D_;   // 8388608
  const size_t DD  = (size_t)D_ * D_;        // 1048576
  __bf16* p   = (__bf16*)d_ws;
  __bf16* Xq  = p; p += BLD;
  __bf16* Xk  = p; p += BLD;
  __bf16* Xv  = p; p += BLD;
  __bf16* Wqb = p; p += DD;
  __bf16* Wkb = p; p += DD;
  __bf16* Wvb = p; p += DD;
  __bf16* Wob = p; p += DD;
  __bf16* Qws = p; p += BLD;
  __bf16* Kws = p; p += BLD;
  __bf16* Vws = p; p += BLD;
  __bf16* Ows = p; p += BLD;

  const int act4 = (int)(BLD / 4), w4 = (int)(DD / 4);
  cvt_f32_bf16_kernel<<<act4 / 256, 256, 0, stream>>>(q_in, Xq, act4);
  cvt_f32_bf16_kernel<<<act4 / 256, 256, 0, stream>>>(k_in, Xk, act4);
  cvt_f32_bf16_kernel<<<act4 / 256, 256, 0, stream>>>(v_in, Xv, act4);
  cvt_f32_bf16_kernel<<<w4 / 256, 256, 0, stream>>>(Wq, Wqb, w4);
  cvt_f32_bf16_kernel<<<w4 / 256, 256, 0, stream>>>(Wk, Wkb, w4);
  cvt_f32_bf16_kernel<<<w4 / 256, 256, 0, stream>>>(Wv, Wvb, w4);
  cvt_f32_bf16_kernel<<<w4 / 256, 256, 0, stream>>>(Wo, Wob, w4);

  dim3 g1(B_ * L_ / 16, 2 * H_, 3);       // 512 x 32 x 3
  qkv_rope_kernel<<<g1, 32, 0, stream>>>(Xq, Xk, Xv, Wqb, Wkb, Wvb,
                                         bq, bk, bv, cosT, sinT,
                                         Qws, Kws, Vws);

  dim3 g2(L_ / 128, H_, B_);              // 16 x 16 x 4, 8 waves/block
  flash_attn_kernel<<<g2, 256, 0, stream>>>(Qws, Kws, Vws, Ows, causal);

  dim3 g3(B_ * L_ / 16, D_ / 16);         // 512 x 64
  out_proj_kernel<<<g3, 32, 0, stream>>>(Ows, Wob, bo, (float*)d_out);
}
